// BaseBidirectionalAttention_82463372083634
// MI455X (gfx1250) — matrix operand order repaired and checked by reading, not compiled, not run on hardware
//
#include <hip/hip_runtime.h>
#include <hip/hip_bf16.h>

// ---------------------------------------------------------------------------
// BiDAF attention + FFN for MI455X (gfx1250), wave32, WMMA bf16 path.
// ---------------------------------------------------------------------------

#define B_  8
#define C_  1024
#define Q_  128
#define E_  512
#define H_  1024
#define E4_ 2048

typedef __bf16 bf16;
typedef __attribute__((ext_vector_type(4)))  bf16  v4bf;
typedef __attribute__((ext_vector_type(8)))  bf16  v8bf;
typedef __attribute__((ext_vector_type(16))) bf16  v16bf;
typedef __attribute__((ext_vector_type(8)))  float v8f;

// ---- WMMA fragment helpers (layouts per CDNA5 ISA 7.12.2) ------------------

// A (bf16 16x32, M x K): lane m = lane&15; hi = lane>>4.
__device__ __forceinline__ v16bf frag_a(const bf16* __restrict__ As, int lane) {
  int m = lane & 15, hi = lane >> 4;
  const bf16* r = As + m * 32;
  v8bf x0 = *(const v8bf*)(r + hi * 8);
  v8bf x1 = *(const v8bf*)(r + 16 + hi * 8);
  return __builtin_shufflevector(x0, x1, 0,1,2,3,4,5,6,7,8,9,10,11,12,13,14,15);
}

// B (bf16 32x16, K x N) stored N-major in LDS as Bs[n][k].
__device__ __forceinline__ v16bf frag_b(const bf16* __restrict__ Bs, int lane) {
  int n = lane & 15, hi = lane >> 4;
  const bf16* r = Bs + n * 32;
  v8bf x0 = *(const v8bf*)(r + hi * 8);
  v8bf x1 = *(const v8bf*)(r + 16 + hi * 8);
  return __builtin_shufflevector(x0, x1, 0,1,2,3,4,5,6,7,8,9,10,11,12,13,14,15);
}

#define WMMA_BF16(a, b, c) \
  __builtin_amdgcn_wmma_f32_16x16x32_bf16(false, (a), false, (b), (short)0, (c), false, false)

__device__ __forceinline__ v4bf cvt4(float4 f) {
  v4bf o; o[0] = (bf16)f.x; o[1] = (bf16)f.y; o[2] = (bf16)f.z; o[3] = (bf16)f.w;
  return o;
}

// ---- wave reductions (wave32) ----------------------------------------------
__device__ __forceinline__ float wsum(float v) {
  #pragma unroll
  for (int off = 16; off; off >>= 1) v += __shfl_xor(v, off, 32);
  return v;
}
__device__ __forceinline__ float wmax(float v) {
  #pragma unroll
  for (int off = 16; off; off >>= 1) v = fmaxf(v, __shfl_xor(v, off, 32));
  return v;
}

// ===========================================================================
// K0: q_w = question . w_question  (B*Q dots), c_w = context . w_context (B*C)
// One wave per dot product of length E=512; float4 loads.
// ===========================================================================
__global__ void k_rowdots(const float* __restrict__ question,
                          const float* __restrict__ context,
                          const float* __restrict__ wq,
                          const float* __restrict__ wc,
                          float* __restrict__ qw, float* __restrict__ cw) {
  int g = blockIdx.x * 8 + (threadIdx.x >> 5);
  int lane = threadIdx.x & 31;
  const float* vec; const float* w; float* out;
  if (g < B_ * Q_) { vec = question + (size_t)g * E_; w = wq; out = qw + g; }
  else { int g2 = g - B_ * Q_; vec = context + (size_t)g2 * E_; w = wc; out = cw + g2; }
  const float4* v4 = (const float4*)vec;
  const float4* w4 = (const float4*)w;
  float s = 0.f;
  #pragma unroll
  for (int e = 0; e < E_ / 128; ++e) {
    float4 a = v4[lane + e * 32], b = w4[lane + e * 32];
    s += a.x * b.x + a.y * b.y + a.z * b.z + a.w * b.w;
  }
  s = wsum(s);
  if (lane == 0) *out = s;
}

// ===========================================================================
// K1: sim = context @ (question*w_mult)^T + q_w + c_w ; softmax over q ;
//     store P (bf16, B*C*Q) and rowmax (B*C).
// Block: 512 threads (16 waves, 4x4), 128x128 tile, wave tile 32x32, K=E=512.
// Sim tile (64KB f32) shares LDS with A/B staging tiles.
// ===========================================================================
__global__ void k_sim_softmax(const float* __restrict__ context,
                              const float* __restrict__ question,
                              const float* __restrict__ wmul,
                              const float* __restrict__ qw,
                              const float* __restrict__ cw,
                              bf16* __restrict__ P,
                              float* __restrict__ rowmax) {
  __shared__ __align__(16) char smem[128 * 128 * 4];   // 64 KB
  bf16*  As  = (bf16*)smem;                            // 128x32 = 8 KB
  bf16*  Bs  = (bf16*)(smem + 8192);                   // 128x32 = 8 KB
  float* Sim = (float*)smem;                           // 128x128, after K loop

  int b = blockIdx.x >> 3, cblk = blockIdx.x & 7;
  int tid = threadIdx.x, lane = tid & 31, wid = tid >> 5;
  int wm = wid >> 2, wn = wid & 3;

  v8f acc[2][2] = {};
  const float* ctx = context + ((size_t)b * C_ + cblk * 128) * E_;
  const float* qst = question + (size_t)b * Q_ * E_;

  for (int k0 = 0; k0 < E_; k0 += 32) {
    #pragma unroll
    for (int j = 0; j < 2; ++j) {               // A: context 128x32, float4
      int fi = j * 512 + tid, r = fi >> 3, c4 = (fi & 7) * 4;
      float4 f = *(const float4*)(ctx + (size_t)r * E_ + k0 + c4);
      *(v4bf*)(As + r * 32 + c4) = cvt4(f);
    }
    #pragma unroll
    for (int j = 0; j < 2; ++j) {               // B: (question*wmul), [q][k]
      int fi = j * 512 + tid, q = fi >> 3, c4 = (fi & 7) * 4;
      float4 f = *(const float4*)(qst + (size_t)q * E_ + k0 + c4);
      float4 m = *(const float4*)(wmul + k0 + c4);
      f.x *= m.x; f.y *= m.y; f.z *= m.z; f.w *= m.w;
      *(v4bf*)(Bs + q * 32 + c4) = cvt4(f);
    }
    __syncthreads();
    #pragma unroll
    for (int i = 0; i < 2; ++i) {
      v16bf a = frag_a(As + (wm * 32 + i * 16) * 32, lane);
      #pragma unroll
      for (int j = 0; j < 2; ++j) {
        v16bf bb = frag_b(Bs + (wn * 32 + j * 16) * 32, lane);
        acc[i][j] = WMMA_BF16(a, bb, acc[i][j]);
      }
    }
    __syncthreads();
  }

  // sim tile -> LDS (+ row/col bias)
  int col = lane & 15, hi = lane >> 4;
  #pragma unroll
  for (int i = 0; i < 2; ++i)
    #pragma unroll
    for (int j = 0; j < 2; ++j) {
      int n = wn * 32 + j * 16 + col;
      #pragma unroll
      for (int r = 0; r < 8; ++r) {
        int m = wm * 32 + i * 16 + r + 8 * hi;
        Sim[m * 128 + n] = acc[i][j][r] + qw[b * Q_ + n] + cw[b * C_ + cblk * 128 + m];
      }
    }
  __syncthreads();

  // row softmax over q (128 wide): 8 rows per wave
  for (int r0 = 0; r0 < 8; ++r0) {
    int m = wid * 8 + r0;
    float v0 = Sim[m * 128 + lane],      v1 = Sim[m * 128 + 32 + lane];
    float v2 = Sim[m * 128 + 64 + lane], v3 = Sim[m * 128 + 96 + lane];
    float mx = wmax(fmaxf(fmaxf(v0, v1), fmaxf(v2, v3)));
    float e0 = __expf(v0 - mx), e1 = __expf(v1 - mx);
    float e2 = __expf(v2 - mx), e3 = __expf(v3 - mx);
    float inv = 1.0f / wsum(e0 + e1 + e2 + e3);
    size_t base = ((size_t)b * C_ + cblk * 128 + m) * Q_;
    P[base + lane]      = (bf16)(e0 * inv);
    P[base + 32 + lane] = (bf16)(e1 * inv);
    P[base + 64 + lane] = (bf16)(e2 * inv);
    P[base + 96 + lane] = (bf16)(e3 * inv);
    if (lane == 0) rowmax[(size_t)b * C_ + cblk * 128 + m] = mx;
  }
}

// ===========================================================================
// K2: q2c[b,e] = softmax_c(rowmax[b,:]) . context[b,:,e]   (one block per b)
// ===========================================================================
__global__ void k_q2c(const float* __restrict__ context,
                      const float* __restrict__ rowmax,
                      float* __restrict__ q2c) {
  __shared__ float w[C_];
  __shared__ float red[8];
  int b = blockIdx.x, tid = threadIdx.x, lane = tid & 31, wid = tid >> 5;

  float mx = -3.0e38f;
  for (int c = tid; c < C_; c += 256) mx = fmaxf(mx, rowmax[b * C_ + c]);
  mx = wmax(mx);
  if (lane == 0) red[wid] = mx;
  __syncthreads();
  mx = red[0];
  #pragma unroll
  for (int i = 1; i < 8; ++i) mx = fmaxf(mx, red[i]);
  __syncthreads();

  float s = 0.f;
  for (int c = tid; c < C_; c += 256) { float e = __expf(rowmax[b * C_ + c] - mx); w[c] = e; s += e; }
  s = wsum(s);
  if (lane == 0) red[wid] = s;
  __syncthreads();
  s = 0.f;
  #pragma unroll
  for (int i = 0; i < 8; ++i) s += red[i];
  float inv = 1.f / s;
  for (int c = tid; c < C_; c += 256) w[c] *= inv;
  __syncthreads();

  for (int e = tid; e < E_; e += 256) {
    float acc = 0.f;
    for (int c = 0; c < C_; ++c) acc += w[c] * context[((size_t)b * C_ + c) * E_ + e];
    q2c[b * E_ + e] = acc;
  }
}

// ===========================================================================
// K3: c2q = P (C x Q, bf16) @ question (Q x E).
// Block tile 128x128, 8 waves (4x2), wave tile 32x64, K=Q=128.
// ===========================================================================
__global__ void k_c2q(const bf16* __restrict__ P,
                      const float* __restrict__ question,
                      float* __restrict__ c2q) {
  __shared__ __align__(16) bf16 As[128 * 32];
  __shared__ __align__(16) bf16 Bs[128 * 32];
  int bi = blockIdx.x;
  int nblk = bi & 3, cblk = (bi >> 2) & 7, b = bi >> 5;
  int tid = threadIdx.x, lane = tid & 31, wid = tid >> 5;
  int wm = wid >> 1, wn = wid & 1;
  int m0 = cblk * 128, n0 = nblk * 128;

  v8f acc[2][4] = {};
  for (int k0 = 0; k0 < Q_; k0 += 32) {
    #pragma unroll
    for (int j = 0; j < 2; ++j) {                // A: P bf16, uint4 copy (8 bf16)
      int fi = j * 256 + tid, r = fi >> 2, c8 = (fi & 3) * 8;
      *(uint4*)(As + r * 32 + c8) =
          *(const uint4*)(P + ((size_t)b * C_ + m0 + r) * Q_ + k0 + c8);
    }
    #pragma unroll
    for (int j = 0; j < 4; ++j) {                // B: question[k][n4] float4 -> Bs[n][k]
      int fi = j * 256 + tid, k = fi >> 5, n4 = (fi & 31) * 4;
      float4 f = *(const float4*)(question + ((size_t)b * Q_ + k0 + k) * E_ + n0 + n4);
      Bs[(n4 + 0) * 32 + k] = (bf16)f.x;
      Bs[(n4 + 1) * 32 + k] = (bf16)f.y;
      Bs[(n4 + 2) * 32 + k] = (bf16)f.z;
      Bs[(n4 + 3) * 32 + k] = (bf16)f.w;
    }
    __syncthreads();
    v16bf bfrag[4];
    #pragma unroll
    for (int j = 0; j < 4; ++j) bfrag[j] = frag_b(Bs + (wn * 64 + j * 16) * 32, lane);
    #pragma unroll
    for (int i = 0; i < 2; ++i) {
      v16bf a = frag_a(As + (wm * 32 + i * 16) * 32, lane);
      #pragma unroll
      for (int j = 0; j < 4; ++j)
        acc[i][j] = WMMA_BF16(a, bfrag[j], acc[i][j]);
    }
    __syncthreads();
  }
  int col = lane & 15, hi = lane >> 4;
  #pragma unroll
  for (int i = 0; i < 2; ++i)
    #pragma unroll
    for (int j = 0; j < 4; ++j)
      #pragma unroll
      for (int r = 0; r < 8; ++r) {
        int m = m0 + wm * 32 + i * 16 + r + 8 * hi;
        int n = n0 + wn * 64 + j * 16 + col;
        c2q[((size_t)b * C_ + m) * E_ + n] = acc[i][j][r];
      }
}

// ===========================================================================
// K4: h = (attended @ W1 + b1) * mask ; attended K-tiles synthesized on the fly:
//   k<512: ctx | k<1024: c2q | k<1536: ctx*c2q | k<2048: ctx*q2c[b]
// M=8192 (b*C+c), N=H=1024, K=2048.  Block tile 128x128, wave tile 32x64.
// Output h in bf16.
// ===========================================================================
__global__ void k_ffn1(const float* __restrict__ context,
                       const float* __restrict__ c2q,
                       const float* __restrict__ q2c,
                       const float* __restrict__ W1,
                       const float* __restrict__ b1,
                       const float* __restrict__ mask,
                       bf16* __restrict__ h) {
  __shared__ __align__(16) bf16 As[128 * 32];
  __shared__ __align__(16) bf16 Bs[128 * 32];
  int bi = blockIdx.x;
  int nblk = bi & 7, mblk = bi >> 3;
  int m0 = mblk * 128, n0 = nblk * 128;
  int tid = threadIdx.x, lane = tid & 31, wid = tid >> 5;
  int wm = wid >> 1, wn = wid & 1;

  v8f acc[2][4] = {};
  for (int k0 = 0; k0 < E4_; k0 += 32) {
    int seg = k0 >> 9, ko = k0 & 511;
    #pragma unroll
    for (int j = 0; j < 4; ++j) {                // A: attended tile, float4
      int fi = j * 256 + tid, r = fi >> 3, c4 = (fi & 7) * 4;
      size_t row = (size_t)(m0 + r);
      size_t off = row * E_ + ko + c4;
      float4 v;
      if (seg == 0) {
        v = *(const float4*)(context + off);
      } else if (seg == 1) {
        v = *(const float4*)(c2q + off);
      } else if (seg == 2) {
        float4 a = *(const float4*)(context + off);
        float4 c = *(const float4*)(c2q + off);
        v.x = a.x * c.x; v.y = a.y * c.y; v.z = a.z * c.z; v.w = a.w * c.w;
      } else {
        float4 a = *(const float4*)(context + off);
        float4 c = *(const float4*)(q2c + (row >> 10) * E_ + ko + c4);
        v.x = a.x * c.x; v.y = a.y * c.y; v.z = a.z * c.z; v.w = a.w * c.w;
      }
      *(v4bf*)(As + r * 32 + c4) = cvt4(v);
    }
    #pragma unroll
    for (int j = 0; j < 4; ++j) {                // B: W1[k][n4] float4 -> Bs[n][k]
      int fi = j * 256 + tid, k = fi >> 5, n4 = (fi & 31) * 4;
      float4 f = *(const float4*)(W1 + (size_t)(k0 + k) * H_ + n0 + n4);
      Bs[(n4 + 0) * 32 + k] = (bf16)f.x;
      Bs[(n4 + 1) * 32 + k] = (bf16)f.y;
      Bs[(n4 + 2) * 32 + k] = (bf16)f.z;
      Bs[(n4 + 3) * 32 + k] = (bf16)f.w;
    }
    if (k0 + 32 < E4_)
      __builtin_prefetch(W1 + (size_t)(k0 + 32) * H_ + n0 + (tid & 127), 0, 1);
    __syncthreads();
    v16bf bfrag[4];
    #pragma unroll
    for (int j = 0; j < 4; ++j) bfrag[j] = frag_b(Bs + (wn * 64 + j * 16) * 32, lane);
    #pragma unroll
    for (int i = 0; i < 2; ++i) {
      v16bf a = frag_a(As + (wm * 32 + i * 16) * 32, lane);
      #pragma unroll
      for (int j = 0; j < 4; ++j)
        acc[i][j] = WMMA_BF16(a, bfrag[j], acc[i][j]);
    }
    __syncthreads();
  }
  int col = lane & 15, hi = lane >> 4;
  #pragma unroll
  for (int i = 0; i < 2; ++i)
    #pragma unroll
    for (int j = 0; j < 4; ++j)
      #pragma unroll
      for (int r = 0; r < 8; ++r) {
        int m = m0 + wm * 32 + i * 16 + r + 8 * hi;
        int n = n0 + wn * 64 + j * 16 + col;
        float hv = (acc[i][j][r] + b1[n]) * mask[m];
        h[(size_t)m * H_ + n] = (bf16)hv;
      }
}

// ===========================================================================
// K5: out = relu((h @ W2 + b2) * mask).  M=8192, N=2048, K=1024. h is bf16.
// Block tile 128x128, wave tile 32x64.
// ===========================================================================
__global__ void k_ffn2(const bf16* __restrict__ h,
                       const float* __restrict__ W2,
                       const float* __restrict__ b2,
                       const float* __restrict__ mask,
                       float* __restrict__ out) {
  __shared__ __align__(16) bf16 As[128 * 32];
  __shared__ __align__(16) bf16 Bs[128 * 32];
  int bi = blockIdx.x;
  int nblk = bi & 15, mblk = bi >> 4;
  int m0 = mblk * 128, n0 = nblk * 128;
  int tid = threadIdx.x, lane = tid & 31, wid = tid >> 5;
  int wm = wid >> 1, wn = wid & 1;

  v8f acc[2][4] = {};
  for (int k0 = 0; k0 < H_; k0 += 32) {
    #pragma unroll
    for (int j = 0; j < 2; ++j) {                // A: h bf16, uint4 copy
      int fi = j * 256 + tid, r = fi >> 2, c8 = (fi & 3) * 8;
      *(uint4*)(As + r * 32 + c8) =
          *(const uint4*)(h + (size_t)(m0 + r) * H_ + k0 + c8);
    }
    #pragma unroll
    for (int j = 0; j < 4; ++j) {                // B: W2[k][n4] float4 -> Bs[n][k]
      int fi = j * 256 + tid, k = fi >> 5, n4 = (fi & 31) * 4;
      float4 f = *(const float4*)(W2 + (size_t)(k0 + k) * E4_ + n0 + n4);
      Bs[(n4 + 0) * 32 + k] = (bf16)f.x;
      Bs[(n4 + 1) * 32 + k] = (bf16)f.y;
      Bs[(n4 + 2) * 32 + k] = (bf16)f.z;
      Bs[(n4 + 3) * 32 + k] = (bf16)f.w;
    }
    if (k0 + 32 < H_)
      __builtin_prefetch(W2 + (size_t)(k0 + 32) * E4_ + n0 + (tid & 127), 0, 1);
    __syncthreads();
    v16bf bfrag[4];
    #pragma unroll
    for (int j = 0; j < 4; ++j) bfrag[j] = frag_b(Bs + (wn * 64 + j * 16) * 32, lane);
    #pragma unroll
    for (int i = 0; i < 2; ++i) {
      v16bf a = frag_a(As + (wm * 32 + i * 16) * 32, lane);
      #pragma unroll
      for (int j = 0; j < 4; ++j)
        acc[i][j] = WMMA_BF16(a, bfrag[j], acc[i][j]);
    }
    __syncthreads();
  }
  int col = lane & 15, hi = lane >> 4;
  #pragma unroll
  for (int i = 0; i < 2; ++i)
    #pragma unroll
    for (int j = 0; j < 4; ++j)
      #pragma unroll
      for (int r = 0; r < 8; ++r) {
        int m = m0 + wm * 32 + i * 16 + r + 8 * hi;
        int n = n0 + wn * 64 + j * 16 + col;
        float o = (acc[i][j][r] + b2[n]) * mask[m];
        out[(size_t)m * E4_ + n] = fmaxf(o, 0.f);
      }
}

// ===========================================================================
extern "C" void kernel_launch(void* const* d_in, const int* in_sizes, int n_in,
                              void* d_out, int out_size, void* d_ws, size_t ws_size,
                              hipStream_t stream) {
  const float* context    = (const float*)d_in[0];
  const float* question   = (const float*)d_in[1];
  const float* mask       = (const float*)d_in[2];
  const float* w_question = (const float*)d_in[3];
  const float* w_context  = (const float*)d_in[4];
  const float* w_multiple = (const float*)d_in[5];
  const float* W1         = (const float*)d_in[6];
  const float* b1         = (const float*)d_in[7];
  const float* W2         = (const float*)d_in[8];
  const float* b2         = (const float*)d_in[9];
  float* out = (float*)d_out;

  char* ws = (char*)d_ws;
  float* qw     = (float*)(ws + 0);                     //  4 KB   (B*Q)
  float* cw     = (float*)(ws + 4096);                  // 32 KB   (B*C)
  float* rowmax = (float*)(ws + 36864);                 // 32 KB   (B*C)
  float* q2c    = (float*)(ws + 69632);                 // 16 KB   (B*E)
  float* c2q    = (float*)(ws + 131072);                // 16 MB   (B*C*E f32)
  bf16*  P      = (bf16*) (ws + 131072 + 16777216);     //  2 MB   (B*C*Q bf16)
  bf16*  h      = (bf16*) (ws + 131072 + 16777216 + 2097152); // 16 MB (B*C*H bf16)

  k_rowdots    <<<dim3((B_ * Q_ + B_ * C_) / 8), dim3(256), 0, stream>>>(question, context, w_question, w_context, qw, cw);
  k_sim_softmax<<<dim3(B_ * (C_ / 128)),         dim3(512), 0, stream>>>(context, question, w_multiple, qw, cw, P, rowmax);
  k_q2c        <<<dim3(B_),                      dim3(256), 0, stream>>>(context, rowmax, q2c);
  k_c2q        <<<dim3(B_ * (C_ / 128) * (E_ / 128)),  dim3(256), 0, stream>>>(P, question, c2q);
  k_ffn1       <<<dim3((B_ * C_ / 128) * (H_ / 128)),  dim3(256), 0, stream>>>(context, c2q, q2c, W1, b1, mask, h);
  k_ffn2       <<<dim3((B_ * C_ / 128) * (E4_ / 128)), dim3(256), 0, stream>>>(h, W2, b2, mask, out);
}
